// POS_MOE_36326833389791
// MI455X (gfx1250) — matrix operand order repaired and checked by reading, not compiled, not run on hardware
//
#include <hip/hip_runtime.h>
#include <hip/hip_bf16.h>

// ---------------------------------------------------------------------------
// Shapes (from reference): B=128, D=20, W=8, E=4, K=1, IN=64, HID=4096
// ---------------------------------------------------------------------------
#define NB   128
#define ND   20
#define NIN  64
#define NHID 4096
#define NE   4

typedef __attribute__((ext_vector_type(16))) __bf16 v16bf;
typedef __attribute__((ext_vector_type(8)))  float  v8f;
typedef __attribute__((ext_vector_type(4)))  unsigned int u32x4;
typedef __attribute__((ext_vector_type(8)))  int  i32x8;
typedef __attribute__((ext_vector_type(4)))  int  i32x4;

#if defined(__HIP_DEVICE_COMPILE__) && defined(__gfx1250__) && \
    __has_builtin(__builtin_amdgcn_tensor_load_to_lds) && \
    __has_builtin(__builtin_amdgcn_s_wait_tensorcnt)
#define HAS_TDM 1
#else
#define HAS_TDM 0
#endif

// ---- fp32 <-> bf16 helpers (RNE), bit-exact ----
__device__ __forceinline__ __bf16 f2bf(float f) {
    unsigned u = __builtin_bit_cast(unsigned, f);
    unsigned r = u + 0x7FFFu + ((u >> 16) & 1u);
    unsigned short h = (unsigned short)(r >> 16);
    return __builtin_bit_cast(__bf16, h);
}
__device__ __forceinline__ float bf2f(__bf16 b) {
    unsigned short s = __builtin_bit_cast(unsigned short, b);
    unsigned u = ((unsigned)s) << 16;
    return __builtin_bit_cast(float, u);
}

#if HAS_TDM
// ---------------------------------------------------------------------------
// TDM: DMA a 32-row x 64-col fp32 tile (row stride = stride_elems) from global
// into LDS at byte offset lds_off, row-major.  Descriptor layout per CDNA5 ISA
// CDNA5 ISA (group0) / §8.4 (group1); groups 2-3 zero (2D).
// This toolchain exposes the 6-arg builtin:
//   (uint32x4 g0, int32x8 g1, int32x4, int32x4, int32x8, i32 cpol)
// ---------------------------------------------------------------------------
__device__ __forceinline__ void tdm_load_tile_32x64_f32(unsigned lds_off,
                                                        const void* gptr,
                                                        unsigned stride_elems) {
    unsigned long long ga = (unsigned long long)(uintptr_t)gptr;
    u32x4 g0;
    g0[0] = 1u;                                            // count=1, user D#
    g0[1] = lds_off;                                       // lds_addr (bytes)
    g0[2] = (unsigned)ga;                                  // global_addr[31:0]
    g0[3] = (unsigned)((ga >> 32) & 0x01FFFFFFu)           // global_addr[56:32]
            | (2u << 30);                                  // type=2 ("image")
    i32x8 g1;
    g1[0] = (int)(2u << 16);                               // data_size=4B; no pad/iter/mcast
    g1[1] = (int)((NHID & 0xFFFFu) << 16);                 // tensor_dim0[15:0]
    g1[2] = (int)((NHID >> 16) | ((NHID & 0xFFFFu) << 16));// dim0[31:16] | dim1[15:0]
    g1[3] = (int)((NHID >> 16) | (64u << 16));             // dim1[31:16] | tile_dim0=64
    g1[4] = (int)32u;                                      // tile_dim1=32, tile_dim2=0
    g1[5] = (int)stride_elems;                             // tensor_dim0_stride[31:0]
    g1[6] = 0;                                             // stride[47:32] | dim1_stride lo
    g1[7] = 0;
    i32x4 gz4 = {0, 0, 0, 0};
    i32x8 gz8 = {0, 0, 0, 0, 0, 0, 0, 0};
    __builtin_amdgcn_tensor_load_to_lds(g0, g1, gz4, gz4, gz8, 0);
}
#endif

// ---------------------------------------------------------------------------
// Kernel 1: conv stack -> flat (B, 64).
// ---------------------------------------------------------------------------
__global__ __launch_bounds__(256)
void conv_kernel(const float* __restrict__ in, const float* __restrict__ w1c,
                 const float* __restrict__ b1c, const float* __restrict__ w2c,
                 const float* __restrict__ b2c, float* __restrict__ flat) {
    __shared__ float r1[10 * ND * 64];
    const int b = blockIdx.x;
    const float* ib = in + (size_t)b * (ND * 64);
    for (int idx = threadIdx.x; idx < 10 * ND * 64; idx += 256) {
        int c = idx / (ND * 64); int rem = idx % (ND * 64);
        int d = rem >> 6; int p = rem & 63; int y = p >> 3; int x = p & 7;
        float acc = b1c[c];
        const float* wc = w1c + c * 27;
        #pragma unroll
        for (int kd = 0; kd < 3; ++kd) {
            int dd = d + kd - 1; if (dd < 0 || dd >= ND) continue;
            #pragma unroll
            for (int ky = 0; ky < 3; ++ky) {
                int yy = y + ky - 1; if (yy < 0 || yy >= 8) continue;
                #pragma unroll
                for (int kx = 0; kx < 3; ++kx) {
                    int xx = x + kx - 1; if (xx < 0 || xx >= 8) continue;
                    acc += ib[dd * 64 + yy * 8 + xx] * wc[kd * 9 + ky * 3 + kx];
                }
            }
        }
        r1[idx] = fmaxf(acc, 0.0f);
    }
    __syncthreads();
    if (threadIdx.x < 64) {
        int y = threadIdx.x >> 3, x = threadIdx.x & 7;
        float acc = b2c[0];
        for (int c = 0; c < 10; ++c)
            for (int d = 0; d < ND; ++d) {
                const float* wc = w2c + (c * ND + d) * 9;
                const float* rc = r1 + (c * ND + d) * 64;
                #pragma unroll
                for (int ky = 0; ky < 3; ++ky) {
                    int yy = y + ky - 1; if (yy < 0 || yy >= 8) continue;
                    #pragma unroll
                    for (int kx = 0; kx < 3; ++kx) {
                        int xx = x + kx - 1; if (xx < 0 || xx >= 8) continue;
                        acc += rc[yy * 8 + xx] * wc[ky * 3 + kx];
                    }
                }
            }
        flat[b * NIN + threadIdx.x] = fmaxf(acc, 0.0f);
    }
}

// ---------------------------------------------------------------------------
// Kernel 2: top-1 gating (gate value == 1.0), routing tables, aux loss.
// ---------------------------------------------------------------------------
__global__ __launch_bounds__(128)
void gate_kernel(const float* __restrict__ flat, const float* __restrict__ wg,
                 int* __restrict__ counts, int* __restrict__ offsets,
                 int* __restrict__ pos2sample, int* __restrict__ sample2pos,
                 float* __restrict__ aux) {
    __shared__ int cnt[NE];
    __shared__ int off[NE];
    const int b = threadIdx.x;
    if (b < NE) cnt[b] = 0;
    __syncthreads();
    float l[NE] = {0.f, 0.f, 0.f, 0.f};
    for (int i = 0; i < NIN; ++i) {
        float f = flat[b * NIN + i];
        #pragma unroll
        for (int e = 0; e < NE; ++e) l[e] += f * wg[i * NE + e];
    }
    int best = 0; float bv = l[0];
    #pragma unroll
    for (int e = 1; e < NE; ++e) if (l[e] > bv) { bv = l[e]; best = e; }
    int p = atomicAdd(&cnt[best], 1);
    __syncthreads();
    if (b == 0) {
        int s = 0;
        for (int e = 0; e < NE; ++e) { off[e] = s; offsets[e] = s; counts[e] = cnt[e]; s += cnt[e]; }
        const float m = (float)NB / (float)NE;
        float var = 0.f;
        for (int e = 0; e < NE; ++e) { float d = (float)cnt[e] - m; var += d * d; }
        var /= (float)(NE - 1);
        aux[0] = 2.0f * (var / (m * m + 1e-10f)) * 1e-2f;
    }
    __syncthreads();
    int pos = off[best] + p;
    pos2sample[pos] = b;
    sample2pos[b]   = pos;
}

// ---------------------------------------------------------------------------
// Kernel 3/4: routed GEMM.  Wave = 16x16 output tile, block = 4 waves = 64
// cols, 16 rows.  Weight tiles (32K x 64N fp32) are DMA'd into LDS by the TDM
// (double-buffered, s_wait_tensorcnt-pipelined); A rows are loaded per-lane
// from global (clamped, scaled by 0 for pad rows -> no exec branching).
// fp32 accuracy via bf16 hi/lo split: Ah*Bh + Al*Bh + Ah*Bl (3 WMMAs/K-tile).
// ---------------------------------------------------------------------------
__global__ __launch_bounds__(128)
void expert_gemm(const float* __restrict__ A, int lda,
                 const float* __restrict__ Bw, int Kdim,
                 const float* __restrict__ bias,
                 float* __restrict__ Out,
                 const int* __restrict__ counts,
                 const int* __restrict__ offsets,
                 const int* __restrict__ rowmap,   // null => identity
                 int use_rowmap, int do_relu) {
    __shared__ float bufs[2][32 * 64];               // double-buffered B tile
    const int e       = blockIdx.z;
    const int cnt     = counts[e];
    const int rowBase = blockIdx.y * 16;
    if (rowBase >= cnt) return;
    const int off  = offsets[e];
    const int wave = threadIdx.x >> 5;
    const int lane = threadIdx.x & 31;
    const int hl   = lane >> 4;
    const int n    = lane & 15;
    const int m    = lane & 15;

    // A row for this lane's A operand: clamp + scale (unconditional loads)
    const int   aPos   = rowBase + m;
    const float aScale = (aPos < cnt) ? 1.0f : 0.0f;
    int aRow = off + ((aPos < cnt) ? aPos : (cnt - 1));
    if (use_rowmap) aRow = rowmap[aRow];
    const float* Ap = A + (size_t)aRow * (size_t)lda;

    // block's 64-wide column slab of W[e]
    const float* Bblk = Bw + (size_t)e * (size_t)Kdim * NHID + blockIdx.x * 64;

    v8f acc = {0.f, 0.f, 0.f, 0.f, 0.f, 0.f, 0.f, 0.f};

#if HAS_TDM
    if (wave == 0)
        tdm_load_tile_32x64_f32((unsigned)(uintptr_t)&bufs[0][0], Bblk, NHID);
#endif
    int cur = 0;
    for (int kt = 0; kt < Kdim; kt += 32) {
#if HAS_TDM
        if (wave == 0) {
            if (kt + 32 < Kdim) {
                tdm_load_tile_32x64_f32((unsigned)(uintptr_t)&bufs[cur ^ 1][0],
                                        Bblk + (size_t)(kt + 32) * NHID, NHID);
                __builtin_amdgcn_s_wait_tensorcnt(1);  // current tile complete
            } else {
                __builtin_amdgcn_s_wait_tensorcnt(0);
            }
        }
        __syncthreads();
#else
        for (int idx = threadIdx.x; idx < 32 * 64; idx += 128)
            bufs[cur][idx] = Bblk[(size_t)(kt + (idx >> 6)) * NHID + (idx & 63)];
        __syncthreads();
#endif
        const float* bt = &bufs[cur][wave * 16 + n];

        v16bf ah, al, bh, bl;
        #pragma unroll
        for (int v = 0; v < 8; ++v) {
            // ---- A: 16x32 bf16 operand (ISA 7.12.2 16-bit A layout) ----
            int ka = kt + ((v & 4) << 2) + hl * 8 + ((v & 3) << 1);
            float a0 = Ap[ka] * aScale;
            float a1 = Ap[ka + 1] * aScale;
            __bf16 h0 = f2bf(a0); ah[2 * v]     = h0; al[2 * v]     = f2bf(a0 - bf2f(h0));
            __bf16 h1 = f2bf(a1); ah[2 * v + 1] = h1; al[2 * v + 1] = f2bf(a1 - bf2f(h1));
            // ---- B: 32x16 bf16 operand, from the LDS-staged tile ----
            int kb = ((v & 4) << 2) + hl * 8 + ((v & 3) << 1);
            float b0 = bt[kb * 64];
            float b1 = bt[kb * 64 + 64];
            __bf16 g0 = f2bf(b0); bh[2 * v]     = g0; bl[2 * v]     = f2bf(b0 - bf2f(g0));
            __bf16 g1 = f2bf(b1); bh[2 * v + 1] = g1; bl[2 * v + 1] = f2bf(b1 - bf2f(g1));
        }

        acc = __builtin_amdgcn_wmma_f32_16x16x32_bf16(false, ah, false, bh, (short)0, acc, false, false);
        acc = __builtin_amdgcn_wmma_f32_16x16x32_bf16(false, al, false, bh, (short)0, acc, false, false);
        acc = __builtin_amdgcn_wmma_f32_16x16x32_bf16(false, ah, false, bl, (short)0, acc, false, false);

        __syncthreads();
        cur ^= 1;
    }

    const int col = blockIdx.x * 64 + wave * 16 + n;
    const float bv = bias[e * NHID + col];
    #pragma unroll
    for (int r = 0; r < 8; ++r) {
        int mm = hl * 8 + r;
        if (rowBase + mm < cnt) {
            float val = acc[r] + bv;
            if (do_relu) val = fmaxf(val, 0.f);
            Out[(size_t)(off + rowBase + mm) * NHID + col] = val;
        }
    }
}

// ---------------------------------------------------------------------------
// Kernel 5: in-place row softmax over 4096.
// ---------------------------------------------------------------------------
__global__ __launch_bounds__(256)
void softmax_kernel(float* __restrict__ o) {
    __shared__ float red[256];
    float* p = o + (size_t)blockIdx.x * NHID;
    float mx = -3.402823466e38f;
    for (int i = threadIdx.x; i < NHID; i += 256) mx = fmaxf(mx, p[i]);
    red[threadIdx.x] = mx; __syncthreads();
    for (int s = 128; s > 0; s >>= 1) {
        if (threadIdx.x < s) red[threadIdx.x] = fmaxf(red[threadIdx.x], red[threadIdx.x + s]);
        __syncthreads();
    }
    mx = red[0]; __syncthreads();
    float sum = 0.f;
    for (int i = threadIdx.x; i < NHID; i += 256) { float v = __expf(p[i] - mx); p[i] = v; sum += v; }
    red[threadIdx.x] = sum; __syncthreads();
    for (int s = 128; s > 0; s >>= 1) {
        if (threadIdx.x < s) red[threadIdx.x] += red[threadIdx.x + s];
        __syncthreads();
    }
    float inv = 1.0f / red[0];
    for (int i = threadIdx.x; i < NHID; i += 256) p[i] *= inv;
}

// ---------------------------------------------------------------------------
// Kernel 6: per-sample 64x64 transform on input + sigmoid.
// ---------------------------------------------------------------------------
__global__ __launch_bounds__(256)
void transform_kernel(const float* __restrict__ in, const float* __restrict__ o,
                      const int* __restrict__ sample2pos, float* __restrict__ out) {
    __shared__ float ys[NHID];
    __shared__ float xin[ND * 64];
    const int b = blockIdx.x;
    const int pos = sample2pos[b];
    const float* yr = o + (size_t)pos * NHID;
    for (int i = threadIdx.x; i < NHID; i += 256) ys[i] = yr[i];
    for (int i = threadIdx.x; i < ND * 64; i += 256) xin[i] = in[(size_t)b * (ND * 64) + i];
    __syncthreads();
    for (int t = threadIdx.x; t < ND * 64; t += 256) {
        int d = t >> 6, i = t & 63;
        const float* yrow = ys + i * 64;
        const float* xrow = xin + d * 64;
        float acc = 0.f;
        #pragma unroll 8
        for (int j = 0; j < 64; ++j) acc += yrow[j] * xrow[j];
        out[(size_t)b * (ND * 64) + t] = 1.0f / (1.0f + __expf(-acc));
    }
}

// ---------------------------------------------------------------------------
// Launcher.  d_in: input, conv1_w, conv1_b, conv2_w, conv2_b, w_gate,
//                  w1, b1, w2, b2.   d_out: out (128*20*64 f32) ++ aux (1 f32)
// ---------------------------------------------------------------------------
extern "C" void kernel_launch(void* const* d_in, const int* in_sizes, int n_in,
                              void* d_out, int out_size, void* d_ws, size_t ws_size,
                              hipStream_t stream) {
    const float* input   = (const float*)d_in[0];
    const float* conv1_w = (const float*)d_in[1];
    const float* conv1_b = (const float*)d_in[2];
    const float* conv2_w = (const float*)d_in[3];
    const float* conv2_b = (const float*)d_in[4];
    const float* w_gate  = (const float*)d_in[5];
    const float* w1      = (const float*)d_in[6];
    const float* b1      = (const float*)d_in[7];
    const float* w2      = (const float*)d_in[8];
    const float* b2      = (const float*)d_in[9];
    float* outp = (float*)d_out;

    float* ws   = (float*)d_ws;
    float* flat = ws;                         // 128*64
    float* h    = flat + NB * NIN;            // 128*4096
    float* o    = h + (size_t)NB * NHID;      // 128*4096
    int*  ibuf       = (int*)(o + (size_t)NB * NHID);
    int*  counts     = ibuf;                  // 4
    int*  offsets    = ibuf + NE;             // 4
    int*  pos2sample = ibuf + 2 * NE;         // 128
    int*  sample2pos = pos2sample + NB;       // 128

    conv_kernel<<<NB, 256, 0, stream>>>(input, conv1_w, conv1_b, conv2_w, conv2_b, flat);
    gate_kernel<<<1, NB, 0, stream>>>(flat, w_gate, counts, offsets, pos2sample,
                                      sample2pos, outp + NB * ND * 64);
    dim3 grid(NHID / 64, NB / 16, NE), block(128);
    expert_gemm<<<grid, block, 0, stream>>>(flat, NIN, w1, NIN, b1, h,
                                            counts, offsets, pos2sample, 1, 1);
    expert_gemm<<<grid, block, 0, stream>>>(h, NHID, w2, NHID, b2, o,
                                            counts, offsets, nullptr, 0, 0);
    softmax_kernel<<<NB, 256, 0, stream>>>(o);
    transform_kernel<<<NB, 256, 0, stream>>>(input, o, sample2pos, outp);
}
